// GraphAttentionNeighbourNetwork_9079560864451
// MI455X (gfx1250) — matrix operand order repaired and checked by reading, not compiled, not run on hardware
//
#include <hip/hip_runtime.h>

// GraphAttentionNeighbourNetwork for MI455X (gfx1250), wave32 + WMMA f16.
// N=4096, NFEAT=512, NHID=64, NCLASS=16, NHEADS=4, NORDERS=2, NIDX=1024.

#define NN      4096
#define NFEAT   512
#define NHID    64
#define NCLASS  16
#define NHEADS  4
#define WORDS   128          // NN / 32
#define NIDX    1024
#define ALPHA   0.2f
#define NEG_BIG (-9.0e15f)

typedef _Float16 f16;
typedef __attribute__((ext_vector_type(16))) _Float16 v16h;
typedef __attribute__((ext_vector_type(8)))  _Float16 v8h;
typedef __attribute__((ext_vector_type(8)))  float    v8f;

// ---------------------------------------------------------------- bit masks
// mask[i][w] bit b  <=>  adj[i][w*32+b] != 0
__global__ void k_pack(const float* __restrict__ adj, unsigned* __restrict__ mask) {
    int t = blockIdx.x * blockDim.x + threadIdx.x;
    if (t >= NN * WORDS) return;
    int row = t / WORDS, w = t % WORDS;
    const float* p = adj + (size_t)row * NN + (size_t)w * 32;
    unsigned m = 0;
#pragma unroll
    for (int b = 0; b < 32; ++b) m |= (p[b] != 0.0f) ? (1u << b) : 0u;
    mask[t] = m;
}

// dst[i] = OR over {k : sel[i] bit k set} of src[k]   (boolean mat-mat product)
__global__ void k_expand(const unsigned* __restrict__ sel,
                         const unsigned* __restrict__ src,
                         unsigned* __restrict__ dst) {
    __shared__ unsigned selrow[WORDS];
    int row = blockIdx.x;
    int tid = threadIdx.x;                    // blockDim == WORDS == 128
    selrow[tid] = sel[(size_t)row * WORDS + tid];
    __syncthreads();
    unsigned acc = 0;
    for (int k = 0; k < NN; ++k) {
        if (selrow[k >> 5] & (1u << (k & 31)))      // uniform across block
            acc |= src[(size_t)k * WORDS + tid];    // coalesced
    }
    dst[(size_t)row * WORDS + tid] = acc;
}

// ---------------------------------------------------------------- converts
__global__ void k_cvt_f16(const float* __restrict__ s, f16* __restrict__ d, int n) {
    int t = blockIdx.x * blockDim.x + threadIdx.x;
    if (t < n) d[t] = (f16)s[t];
}

// src f32 [R][C]  ->  dst f16 [C][R]
__global__ void k_transpose_cvt(const float* __restrict__ src, f16* __restrict__ dst,
                                int R, int C) {
    int t = blockIdx.x * blockDim.x + threadIdx.x;
    if (t >= R * C) return;
    int r = t / C, c = t % C;
    dst[(size_t)c * R + r] = (f16)src[t];
}

// ---------------------------------------------------------------- WMMA GEMM
// C[M][ldc] (f32) = A[M][K] (f16 row-major) x Bt[NT*16][K] (f16, B stored transposed).
// One wave computes a 16-row x (NT*16)-col tile. blockDim = 256 (8 waves).
// NT is compile-time so the accumulators stay in fixed VGPRs (no v_movrel).
template <int NT>
__global__ void __launch_bounds__(256)
k_gemm(const f16* __restrict__ A, const f16* __restrict__ Bt, float* __restrict__ C,
       int M, int K, int ldc) {
    int wave = threadIdx.x >> 5;
    int lane = threadIdx.x & 31;
    int i0 = (blockIdx.x * 8 + wave) * 16;
    if (i0 >= M) return;                       // uniform per wave
    int  n16 = lane & 15;
    bool hi  = lane >= 16;
    const f16* arow = A + (size_t)(i0 + n16) * K;
    v8f acc[NT];
#pragma unroll
    for (int t = 0; t < NT; ++t)
#pragma unroll
        for (int e = 0; e < 8; ++e) acc[t][e] = 0.0f;
    for (int k0 = 0; k0 < K; k0 += 32) {
        // A fragment: lanes 0-15 hold K {0..7,16..23}, lanes 16-31 {8..15,24..31}
        v8h alo = *(const v8h*)(arow + k0 + (hi ? 8 : 0));
        v8h ahi = *(const v8h*)(arow + k0 + 16 + (hi ? 8 : 0));
        v16h a;
#pragma unroll
        for (int e = 0; e < 8; ++e) { a[e] = alo[e]; a[8 + e] = ahi[e]; }
#pragma unroll
        for (int t = 0; t < NT; ++t) {
            // B fragment: lane = column, lanes 0-15 K {0..7,16..23}, lanes 16-31 {8..15,24..31}
            const f16* bcol = Bt + (size_t)(t * 16 + n16) * K + k0 + (hi ? 8 : 0);
            v8h blo = *(const v8h*)bcol;
            v8h bhi = *(const v8h*)(bcol + 16);
            v16h b;
#pragma unroll
            for (int e = 0; e < 8; ++e) { b[e] = blo[e]; b[8 + e] = bhi[e]; }
            acc[t] = __builtin_amdgcn_wmma_f32_16x16x32_f16(
                false, a, false, b, (short)0, acc[t], false, false);
        }
    }
#pragma unroll
    for (int t = 0; t < NT; ++t)
#pragma unroll
        for (int r = 0; r < 8; ++r)
            C[(size_t)(i0 + r + (hi ? 8 : 0)) * ldc + t * 16 + n16] = acc[t][r];
}

// -------------------------------------------------- s_src / s_dst per row
__global__ void k_rowstats(const float* __restrict__ H, const float* __restrict__ a,
                           float* __restrict__ ssrc, float* __restrict__ sdst, int F) {
    int i = blockIdx.x * blockDim.x + threadIdx.x;
    if (i >= NN) return;
    const float* h = H + (size_t)i * F;
    float s0 = 0.f, s1 = 0.f;
    for (int c = 0; c < F; ++c) { s0 += h[c] * a[c]; s1 += h[c] * a[F + c]; }
    ssrc[i] = s0;
    sdst[i] = s1;
}

// -------------------------------------------------- softmax pass 1: max & denom
__global__ void __launch_bounds__(256)
k_att1(const float* __restrict__ ssrc, const float* __restrict__ sdst,
       const unsigned* __restrict__ mask, float* __restrict__ rowm,
       float* __restrict__ rowd) {
    int wave = threadIdx.x >> 5, lane = threadIdx.x & 31;
    int i = blockIdx.x * 8 + wave;
    float si = ssrc[i];
    const unsigned* mrow = mask + (size_t)i * WORDS;
    float mx = -INFINITY;
    for (int w = 0; w < WORDS; ++w) {
        unsigned word = mrow[w];
        float e = si + sdst[w * 32 + lane];
        float l = e > 0.f ? e : ALPHA * e;
        float v = ((word >> lane) & 1u) ? l : NEG_BIG;
        mx = fmaxf(mx, v);
    }
#pragma unroll
    for (int off = 16; off; off >>= 1) mx = fmaxf(mx, __shfl_xor(mx, off, 32));
    float sum = 0.f;
    for (int w = 0; w < WORDS; ++w) {
        unsigned word = mrow[w];
        float e = si + sdst[w * 32 + lane];
        float l = e > 0.f ? e : ALPHA * e;
        float v = ((word >> lane) & 1u) ? l : NEG_BIG;
        sum += __expf(v - mx);
    }
#pragma unroll
    for (int off = 16; off; off >>= 1) sum += __shfl_xor(sum, off, 32);
    if (lane == 0) { rowm[i] = mx; rowd[i] = sum; }
}

// -------------------------------------------------- softmax pass 2: P @ H via WMMA
// HT: f16 [F][NN] (features transposed).  Out: f32, row i -> Out[i*ldo + colOff + c].
template <int NT, bool ELU>
__global__ void __launch_bounds__(256)
k_att2(const float* __restrict__ ssrc, const float* __restrict__ sdst,
       const float* __restrict__ rowm, const float* __restrict__ rowd,
       const unsigned* __restrict__ mask, const f16* __restrict__ HT,
       float* __restrict__ Out, int ldo, int colOff) {
    int wave = threadIdx.x >> 5, lane = threadIdx.x & 31;
    int i0 = (blockIdx.x * 8 + wave) * 16;
    int  n16 = lane & 15;
    bool hi  = lane >= 16;
    int  iM  = i0 + n16;                 // A-fragment row handled by this lane
    float si    = ssrc[iM];
    float mi    = rowm[iM];
    float invd  = 1.0f / rowd[iM];
    const unsigned* mrow = mask + (size_t)iM * WORDS;
    int kb0 = hi ? 8 : 0;                // K elements 0..7  of fragment
    int kb1 = 16 + kb0;                  // K elements 8..15 of fragment
    v8f acc[NT];
#pragma unroll
    for (int t = 0; t < NT; ++t)
#pragma unroll
        for (int e = 0; e < 8; ++e) acc[t][e] = 0.0f;
    for (int j0 = 0; j0 < NN; j0 += 32) {
        unsigned mw = mrow[j0 >> 5];
        v16h a;
#pragma unroll
        for (int e = 0; e < 8; ++e) {
            int k = kb0 + e;
            float ee = si + sdst[j0 + k];
            float l  = ee > 0.f ? ee : ALPHA * ee;
            float v  = ((mw >> k) & 1u) ? l : NEG_BIG;
            a[e] = (f16)(__expf(v - mi) * invd);
            k = kb1 + e;
            ee = si + sdst[j0 + k];
            l  = ee > 0.f ? ee : ALPHA * ee;
            v  = ((mw >> k) & 1u) ? l : NEG_BIG;
            a[8 + e] = (f16)(__expf(v - mi) * invd);
        }
#pragma unroll
        for (int t = 0; t < NT; ++t) {
            const f16* bcol = HT + (size_t)(t * 16 + n16) * NN + j0 + (hi ? 8 : 0);
            v8h blo = *(const v8h*)bcol;
            v8h bhi = *(const v8h*)(bcol + 16);
            v16h b;
#pragma unroll
            for (int e = 0; e < 8; ++e) { b[e] = blo[e]; b[8 + e] = bhi[e]; }
            acc[t] = __builtin_amdgcn_wmma_f32_16x16x32_f16(
                false, a, false, b, (short)0, acc[t], false, false);
        }
    }
#pragma unroll
    for (int t = 0; t < NT; ++t)
#pragma unroll
        for (int r = 0; r < 8; ++r) {
            float v = acc[t][r];
            if (ELU) v = v > 0.f ? v : (__expf(v) - 1.0f);   // ELU alpha=1
            Out[(size_t)(i0 + r + (hi ? 8 : 0)) * ldo + colOff + t * 16 + n16] = v;
        }
}

// ---------------------------------------------------------------- gather
__global__ void k_gather(const float* __restrict__ x3, const int* __restrict__ idx,
                         float* __restrict__ out) {
    int t = blockIdx.x * blockDim.x + threadIdx.x;
    if (t >= NIDX * NCLASS) return;
    int r = t / NCLASS, c = t % NCLASS;
    out[t] = x3[(size_t)idx[r] * NCLASS + c];
}

// ================================================================ host side
extern "C" void kernel_launch(void* const* d_in, const int* in_sizes, int n_in,
                              void* d_out, int out_size, void* d_ws, size_t ws_size,
                              hipStream_t stream) {
    const float* nodes   = (const float*)d_in[0];    // [4096,512]
    const float* adj     = (const float*)d_in[1];    // [4096,4096]
    const int*   idx     = (const int*)  d_in[2];    // [1024]
    const float* W_heads = (const float*)d_in[3];    // [4,512,64]
    const float* a_heads = (const float*)d_in[4];    // [4,128,1]
    const float* W_out   = (const float*)d_in[5];    // [256,16]
    const float* a_out   = (const float*)d_in[6];    // [32,1]
    float* out = (float*)d_out;                      // [1024,16]
    (void)in_sizes; (void)n_in; (void)out_size; (void)ws_size;

    char* wp = (char*)d_ws;
    auto alloc = [&](size_t bytes) -> char* {
        char* p = wp; wp += (bytes + 255) & ~(size_t)255; return p;
    };
    f16*      x16   = (f16*)     alloc((size_t)NN * NFEAT * 2);       // x as f16
    f16*      Wt    = (f16*)     alloc((size_t)NHEADS * NHID * NFEAT * 2);
    f16*      Wot   = (f16*)     alloc((size_t)NCLASS * (NHID * NHEADS) * 2);
    unsigned* mask1 = (unsigned*)alloc((size_t)NN * WORDS * 4);
    unsigned* mask2 = (unsigned*)alloc((size_t)NN * WORDS * 4);
    unsigned* mask3 = (unsigned*)alloc((size_t)NN * WORDS * 4);
    float*    h1    = (float*)   alloc((size_t)NHEADS * NN * NHID * 4);
    f16*      ht1   = (f16*)     alloc((size_t)NHEADS * NHID * NN * 2);
    float*    ssrc  = (float*)   alloc((size_t)NHEADS * NN * 4);
    float*    sdst  = (float*)   alloc((size_t)NHEADS * NN * 4);
    float*    rm1   = (float*)   alloc((size_t)NHEADS * NN * 4);
    float*    rd1   = (float*)   alloc((size_t)NHEADS * NN * 4);
    float*    x2f   = (float*)   alloc((size_t)NN * NHID * NHEADS * 4);
    f16*      x2h   = (f16*)     alloc((size_t)NN * NHID * NHEADS * 2);
    float*    h2    = (float*)   alloc((size_t)NN * NCLASS * 4);
    f16*      ht2   = (f16*)     alloc((size_t)NCLASS * NN * 2);
    float*    s2s   = (float*)   alloc((size_t)NN * 4);
    float*    s2d   = (float*)   alloc((size_t)NN * 4);
    float*    rm2   = (float*)   alloc((size_t)NN * 4);
    float*    rd2   = (float*)   alloc((size_t)NN * 4);
    float*    x3    = (float*)   alloc((size_t)NN * NCLASS * 4);

    // 1) adjacency bit-masks for A, A^2, A^3 (values irrelevant, only >0 pattern)
    k_pack<<<(NN * WORDS + 255) / 256, 256, 0, stream>>>(adj, mask1);
    k_expand<<<NN, WORDS, 0, stream>>>(mask1, mask1, mask2);   // A^2 pattern
    k_expand<<<NN, WORDS, 0, stream>>>(mask1, mask2, mask3);   // A^3 = A * A^2 pattern

    // 2) precision conversions / transposes
    k_cvt_f16<<<(NN * NFEAT + 255) / 256, 256, 0, stream>>>(nodes, x16, NN * NFEAT);
    for (int hd = 0; hd < NHEADS; ++hd)
        k_transpose_cvt<<<(NFEAT * NHID + 255) / 256, 256, 0, stream>>>(
            W_heads + (size_t)hd * NFEAT * NHID, Wt + (size_t)hd * NHID * NFEAT,
            NFEAT, NHID);
    k_transpose_cvt<<<(NHID * NHEADS * NCLASS + 255) / 256, 256, 0, stream>>>(
        W_out, Wot, NHID * NHEADS, NCLASS);

    const unsigned* masks[3] = {mask1, mask2, mask3};

    // 3) layer 1, per head: h = x@W (WMMA), row stats, softmax, att@h (WMMA) + ELU
    for (int hd = 0; hd < NHEADS; ++hd) {
        float* hh  = h1  + (size_t)hd * NN * NHID;
        f16*   hht = ht1 + (size_t)hd * NHID * NN;
        float* ss  = ssrc + (size_t)hd * NN;
        float* sd  = sdst + (size_t)hd * NN;
        k_gemm<NHID / 16><<<NN / (16 * 8), 256, 0, stream>>>(
            x16, Wt + (size_t)hd * NHID * NFEAT, hh, NN, NFEAT, NHID);
        k_rowstats<<<(NN + 255) / 256, 256, 0, stream>>>(
            hh, a_heads + (size_t)hd * 2 * NHID, ss, sd, NHID);
        k_transpose_cvt<<<(NN * NHID + 255) / 256, 256, 0, stream>>>(hh, hht, NN, NHID);
        const unsigned* mk = masks[hd % 3];   // head i uses mats[i % (NORDERS+1)]
        k_att1<<<NN / 8, 256, 0, stream>>>(ss, sd, mk,
                                           rm1 + (size_t)hd * NN, rd1 + (size_t)hd * NN);
        k_att2<NHID / 16, true><<<NN / (16 * 8), 256, 0, stream>>>(
            ss, sd, rm1 + (size_t)hd * NN, rd1 + (size_t)hd * NN, mk, hht,
            x2f, NHID * NHEADS, hd * NHID);
    }

    // 4) layer 2: h2 = elu(concat)@W_out, softmax over adj, att@h2
    k_cvt_f16<<<(NN * NHID * NHEADS + 255) / 256, 256, 0, stream>>>(
        x2f, x2h, NN * NHID * NHEADS);
    k_gemm<NCLASS / 16><<<NN / (16 * 8), 256, 0, stream>>>(
        x2h, Wot, h2, NN, NHID * NHEADS, NCLASS);
    k_rowstats<<<(NN + 255) / 256, 256, 0, stream>>>(h2, a_out, s2s, s2d, NCLASS);
    k_transpose_cvt<<<(NN * NCLASS + 255) / 256, 256, 0, stream>>>(h2, ht2, NN, NCLASS);
    k_att1<<<NN / 8, 256, 0, stream>>>(s2s, s2d, mask1, rm2, rd2);
    k_att2<NCLASS / 16, false><<<NN / (16 * 8), 256, 0, stream>>>(
        s2s, s2d, rm2, rd2, mask1, ht2, x3, NCLASS, 0);

    // 5) out = x3[idx]
    k_gather<<<(NIDX * NCLASS + 255) / 256, 256, 0, stream>>>(x3, idx, out);
}
